// HolographicEncoder_67731634258095
// MI455X (gfx1250) — matrix operand order repaired and checked by reading, not compiled, not run on hardware
//
#include <hip/hip_runtime.h>

// ---- problem constants (match reference) ----
#define NN    128      // nodes
#define DD    64       // feature dim
#define HIDN  128      // hidden
#define NEVK  16       // eigenvectors kept
#define NSWEEPS 8      // Jacobi sweeps
#define AST   129      // f32 LDS row stride (bank-conflict pad; matches TDM pad cfg)
#define ABST  136      // bf16 adjacency LDS stride (16B-aligned rows)
#define KST   136      // bf16 transposed-feature LDS stride (16B-aligned rows)

typedef __attribute__((ext_vector_type(16))) __bf16 v16bf;
typedef __attribute__((ext_vector_type(8)))  __bf16 v8bf;
typedef __attribute__((ext_vector_type(8)))  float  v8f;
typedef __attribute__((ext_vector_type(4)))  unsigned int v4u;
typedef __attribute__((ext_vector_type(4)))  int    v4i;
typedef __attribute__((ext_vector_type(8)))  int    v8i;

__device__ __forceinline__ __bf16 f2bf(float f) {
  unsigned u = __builtin_bit_cast(unsigned, f);
  u += 0x7FFFu + ((u >> 16) & 1u);             // round-to-nearest-even
  unsigned short h = (unsigned short)(u >> 16);
  return __builtin_bit_cast(__bf16, h);
}
__device__ __forceinline__ float bf2f(__bf16 h) {
  unsigned u = ((unsigned)__builtin_bit_cast(unsigned short, h)) << 16;
  return __builtin_bit_cast(float, u);
}

// ---------------------------------------------------------------------
// TDM: DMA one 128x128 f32 tile (row-major, stride 128) from global into
// LDS with +1 DWORD padding every 128 DWORDs -> LDS rows of 129 floats.
// D# per CDNA5 ISA ch.8: group0 = {count, lds_addr, global_addr, type=2},
// group1 = {data_size=4B, pad_enable, pad_interval=128dw, pad_amount=1dw,
//           tensor_dim0/1=128, tile_dim0/1=128, dim0_stride=128}.
// Issue from ONE wave only (TDM ignores EXEC); caller barriers afterwards.
// ---------------------------------------------------------------------
__device__ __forceinline__ void tdm_load_adj_128x129(const float* gsrc, void* lds_dst) {
  unsigned ldsOff = (unsigned)(unsigned long long)lds_dst;
  unsigned long long ga = (unsigned long long)gsrc;
  v4u g0;
  g0.x = 1u;                                                   // count=1 (valid user D#)
  g0.y = ldsOff;                                               // lds_addr (bytes)
  g0.z = (unsigned)(ga & 0xFFFFFFFFull);                       // global_addr[31:0]
  g0.w = (unsigned)((ga >> 32) & 0x1FFFFFFull) | 0x80000000u;  // global_addr[56:32] | type=2
  v8i g1;
  g1[0] = (int)((2u << 16) | (1u << 20) | (6u << 22));         // data_size=4B, pad_en, every 128dw
  g1[1] = (int)(128u << 16);                                   // tensor_dim0 = 128
  g1[2] = (int)(128u << 16);                                   // tensor_dim1 = 128
  g1[3] = (int)(128u << 16);                                   // tile_dim0   = 128
  g1[4] = (int)128u;                                           // tile_dim1   = 128
  g1[5] = (int)128u;                                           // tensor_dim0_stride = 128
  g1[6] = 0;
  g1[7] = 0;
  v4i z4 = {0, 0, 0, 0};
#if defined(__clang_major__) && (__clang_major__ >= 23)
  v8i z8 = {0, 0, 0, 0, 0, 0, 0, 0};
  __builtin_amdgcn_tensor_load_to_lds(g0, g1, z4, z4, z8, 0);
#else
  __builtin_amdgcn_tensor_load_to_lds(g0, g1, z4, z4, 0);
#endif
}

// =====================================================================
// Kernel A: per-batch normalized Laplacian -> parallel cyclic Jacobi
// eigensolver -> eigen-pooling -> spectral MLP head.
// One workgroup (256 threads = 8 waves) per batch element.
// Hot loop: fixed pair-per-thread mapping (k = t>>2), 3 barriers/step.
// =====================================================================
__global__ __launch_bounds__(256)
void holo_eig_spectral(const float* __restrict__ adj,
                       const float* __restrict__ sp_w1, const float* __restrict__ sp_b1,
                       const float* __restrict__ sp_w2, const float* __restrict__ sp_b2,
                       float* __restrict__ out_spectral, float* __restrict__ out_selvals) {
  __shared__ float As[NN][AST];   // working symmetric matrix (becomes ~diagonal)
  __shared__ float Vs[NN][AST];   // accumulated eigenvectors (columns)
  __shared__ float dis[NN];
  __shared__ int   topi[64], boti[64];
  __shared__ int   pp_[64], qq_[64];
  __shared__ float cs_[64], sn_[64];
  __shared__ float evals[NN];
  __shared__ int   pos[NN];
  __shared__ float selv[NEVK], wts[NEVK], pooled[NEVK], hbuf[HIDN];

  const int t = threadIdx.x;
  const int b = blockIdx.x;
  const int wave = t >> 5;
  const float* Ab = adj + (size_t)b * NN * NN;

  // stage adjacency into padded LDS via the Tensor Data Mover
  if (wave == 0) {
    tdm_load_adj_128x129(Ab, (void*)&As[0][0]);
    __builtin_amdgcn_s_wait_tensorcnt(0);
  }
  __syncthreads();

  // degree -> (deg + 1e-8)^(-1/2)
  if (t < NN) {
    float s = 0.f;
    for (int j = 0; j < NN; ++j) s += As[t][j];
    dis[t] = rsqrtf(s + 1e-8f);
  }
  __syncthreads();

  // L = I - A * d_i * d_j  ;  V = I
  for (int idx = t; idx < NN * NN; idx += 256) {
    int i = idx >> 7, j = idx & 127;
    As[i][j] = ((i == j) ? 1.f : 0.f) - As[i][j] * dis[i] * dis[j];
    Vs[i][j] = (i == j) ? 1.f : 0.f;
  }
  if (t < 64) { topi[t] = 2 * t; boti[t] = 2 * t + 1; }
  __syncthreads();

  // --- parallel cyclic Jacobi, Brent-Luk tournament ordering ---
  const int kfix = t >> 2;        // fixed rotation pair per thread (4 threads/pair)
  const int ifix = t & 3;         // row/col phase within pair

  for (int sweep = 0; sweep < NSWEEPS; ++sweep) {
    for (int step = 0; step < NN - 1; ++step) {
      // 64 disjoint rotations + read next-round pairing (old arrays)
      int ntop = 0, nbot = 0;
      if (t < 64) {
        int p = topi[t], q = boti[t];
        ntop = (t == 0) ? p : ((t == 1) ? boti[0] : topi[t - 1]);
        nbot = (t < 63) ? boti[t + 1] : topi[63];
        if (p > q) { int tmp = p; p = q; q = tmp; }
        pp_[t] = p; qq_[t] = q;
        float app = As[p][p], aqq = As[q][q], apq = As[p][q];
        float c = 1.f, s = 0.f;
        if (fabsf(apq) > 1e-10f) {
          float tau = (aqq - app) / (2.f * apq);
          float tt  = ((tau >= 0.f) ? 1.f : -1.f) / (fabsf(tau) + sqrtf(1.f + tau * tau));
          c = rsqrtf(1.f + tt * tt);
          s = tt * c;
        }
        cs_[t] = c; sn_[t] = s;
      }
      __syncthreads();                                  // (1)

      // publish next-round pairing while column update runs (uses pp_/qq_ only)
      if (t < 64) { topi[t] = ntop; boti[t] = nbot; }

      // column update: A <- A*J and V <- V*J  (pair coeffs hoisted to regs)
      {
        const int p = pp_[kfix], q = qq_[kfix];
        const float c = cs_[kfix], s = sn_[kfix];
        for (int r = 0; r < 32; ++r) {
          const int i = ifix + 4 * r;
          float x = As[i][p], y = As[i][q];
          As[i][p] = c * x - s * y;
          As[i][q] = s * x + c * y;
          x = Vs[i][p]; y = Vs[i][q];
          Vs[i][p] = c * x - s * y;
          Vs[i][q] = s * x + c * y;
        }
      }
      __syncthreads();                                  // (2)

      // row update: A <- J^T*A
      {
        const int p = pp_[kfix], q = qq_[kfix];
        const float c = cs_[kfix], s = sn_[kfix];
        for (int r = 0; r < 32; ++r) {
          const int j = ifix + 4 * r;
          float x = As[p][j], y = As[q][j];
          As[p][j] = c * x - s * y;
          As[q][j] = s * x + c * y;
        }
      }
      __syncthreads();                                  // (3)
    }
  }

  // eigenvalues + rank sort (ascending, index tiebreak)
  if (t < NN) evals[t] = As[t][t];
  __syncthreads();
  if (t < NN) {
    float v = evals[t];
    int r = 0;
    for (int j = 0; j < NN; ++j) {
      float vj = evals[j];
      if (vj < v || (vj == v && j < t)) ++r;
    }
    pos[r] = t;
  }
  __syncthreads();
  if (t < NEVK) selv[t] = evals[pos[t + 1]];   // skip lambda_0
  __syncthreads();

  // w = softmax(-selv)
  if (t == 0) {
    float mx = -selv[0];
    for (int k = 1; k < NEVK; ++k) mx = fmaxf(mx, -selv[k]);
    float sum = 0.f;
    for (int k = 0; k < NEVK; ++k) { float e = expf(-selv[k] - mx); wts[k] = e; sum += e; }
    float inv = 1.f / sum;
    for (int k = 0; k < NEVK; ++k) wts[k] *= inv;
  }
  __syncthreads();

  // pooled[k] = w[k] * sum_i V[i][col_k]
  if (t < NEVK) {
    int col = pos[t + 1];
    float s = 0.f;
    for (int i = 0; i < NN; ++i) s += Vs[i][col];
    pooled[t] = wts[t] * s;
  }
  __syncthreads();

  // h = relu(pooled @ sp_w1 + b1)
  if (t < HIDN) {
    float acc = sp_b1[t];
    for (int k = 0; k < NEVK; ++k) acc += pooled[k] * sp_w1[k * HIDN + t];
    hbuf[t] = fmaxf(acc, 0.f);
  }
  __syncthreads();

  // spectral = h @ sp_w2 + b2
  if (t < DD) {
    float acc = sp_b2[t];
    for (int j = 0; j < HIDN; ++j) acc += hbuf[j] * sp_w2[j * DD + t];
    out_spectral[(size_t)b * DD + t] = acc;
  }
  if (t < NEVK) out_selvals[(size_t)b * NEVK + t] = selv[t];
}

// =====================================================================
// Kernel B: row-stochastic propagation via bf16 WMMA (f32 accumulate),
// weighted multiscale means, output MLP, hyperbolic clipping.
// Feature buffers are TRANSPOSED in LDS (fT[d][k]) so every WMMA B
// fragment / D store is contiguous 16B vector LDS traffic.
// One workgroup (256 threads = 8 waves) per batch element.
// =====================================================================
__global__ __launch_bounds__(256)
void holo_prop_out(const float* __restrict__ adj,  const float* __restrict__ feats,
                   const float* __restrict__ scale_w, const float* __restrict__ curv,
                   const float* __restrict__ ow1, const float* __restrict__ ob1,
                   const float* __restrict__ ow2, const float* __restrict__ ob2,
                   const float* __restrict__ spectral,
                   float* __restrict__ out_z, float* __restrict__ out_ze,
                   float* __restrict__ out_c) {
  __shared__ float  As[NN][AST];                 // staged adjacency f32 (TDM)
  __shared__ alignas(16) __bf16 adjb[NN][ABST];  // row-normalized adjacency, bf16
  __shared__ alignas(16) __bf16 fTa[DD][KST];    // transposed feature ping
  __shared__ alignas(16) __bf16 fTb[DD][KST];    // transposed feature pong
  __shared__ float  degs[NN];
  __shared__ float  meanacc[DD];
  __shared__ float  comb[HIDN];
  __shared__ float  hbuf[HIDN];
  __shared__ float  zebuf[NEVK + 1];

  const int t = threadIdx.x;
  const int b = blockIdx.x;
  const int lane = t & 31;
  const int wave = t >> 5;

  // softmax over the 3 scale weights (cheap, per-thread)
  float s0 = scale_w[0], s1 = scale_w[1], s2 = scale_w[2];
  float mw = fmaxf(s0, fmaxf(s1, s2));
  float e0 = expf(s0 - mw), e1 = expf(s1 - mw), e2 = expf(s2 - mw);
  float einv = 1.f / (e0 + e1 + e2);
  float w0 = e0 * einv, w1 = e1 * einv, w2 = e2 * einv;

  // stage adjacency via TDM; transposed bf16 features while DMA runs
  const float* Ab = adj + (size_t)b * NN * NN;
  const float* Fb = feats + (size_t)b * NN * DD;
  if (wave == 0) {
    tdm_load_adj_128x129(Ab, (void*)&As[0][0]);
  }
  for (int idx = t; idx < NN * DD; idx += 256) {
    int i = idx >> 6, d = idx & 63;
    fTa[d][i] = f2bf(Fb[idx]);                   // transpose on store
  }
  if (t < DD) {                                  // mean0 from f32 global (full precision)
    float s = 0.f;
    for (int i = 0; i < NN; ++i) s += Fb[i * DD + t];
    meanacc[t] = w0 * (s * (1.f / NN));
  }
  if (wave == 0) __builtin_amdgcn_s_wait_tensorcnt(0);
  __syncthreads();

  // degrees + row-normalized adjacency in bf16
  if (t < NN) {
    float s = 0.f;
    for (int j = 0; j < NN; ++j) s += As[t][j];
    degs[t] = s + 1e-8f;
  }
  __syncthreads();
  for (int idx = t; idx < NN * NN; idx += 256) {
    int i = idx >> 7, j = idx & 127;
    adjb[i][j] = f2bf(As[i][j] / degs[i]);
  }
  __syncthreads();

  // --- hoist A fragments (row_adj is constant across both scales) ---
  // 16x32 bf16 A layout: lane 0-15 -> K {kb..kb+7, kb+16..kb+23}, lane 16-31 -> +8
  v16bf afrag[4];
  {
    const int mrow = (wave * 16) + (lane & 15);
    const int koff = (lane & 16) ? 8 : 0;
#pragma unroll
    for (int kt = 0; kt < 4; ++kt) {
      const int kb = kt * 32 + koff;
      v8bf lo = *(const v8bf*)&adjb[mrow][kb];
      v8bf hi = *(const v8bf*)&adjb[mrow][kb + 16];
      afrag[kt] = __builtin_shufflevector(lo, hi, 0,1,2,3,4,5,6,7,8,9,10,11,12,13,14,15);
    }
  }

  // --- two propagation scales: dst = row_adj @ src (128x128 @ 128x64) ---
  __bf16 (*src)[KST] = fTa;
  __bf16 (*dst)[KST] = fTb;
  const float wsArr[2] = { w1, w2 };

  for (int sc = 0; sc < 2; ++sc) {
    const int Mbase = wave * 16;                 // each wave: one 16x64 strip
#pragma unroll
    for (int nt = 0; nt < 4; ++nt) {
      const int ncol = nt * 16 + (lane & 15);
      v8f c = {};
#pragma unroll
      for (int kt = 0; kt < 4; ++kt) {
        // 32x16 bf16 B layout: lanes 0-15 K {kb..kb+7, kb+16..kb+23}, lanes 16-31 -> +8 (matches A)
        const int kb = kt * 32 + ((lane & 16) ? 8 : 0);
        v8bf lo = *(const v8bf*)&src[ncol][kb];
        v8bf hi = *(const v8bf*)&src[ncol][kb + 16];
        v16bf bm = __builtin_shufflevector(lo, hi, 0,1,2,3,4,5,6,7,8,9,10,11,12,13,14,15);
        c = __builtin_amdgcn_wmma_f32_16x16x32_bf16(
              /*neg_a=*/false, afrag[kt], /*neg_b=*/false, bm,
              /*c_mod=*/(short)0, c, /*reuse_a=*/false, /*reuse_b=*/false);
      }
      // D store: VGPR r -> row Mbase + r + (lane>=16 ? 8 : 0); contiguous in fT
      const int rbase = Mbase + ((lane & 16) ? 8 : 0);
      v8bf pk;
#pragma unroll
      for (int r = 0; r < 8; ++r) pk[r] = f2bf(c[r]);
      *(v8bf*)&dst[ncol][rbase] = pk;
    }
    __syncthreads();
    if (t < DD) {
      float s = 0.f;
      for (int i = 0; i < NN; ++i) s += bf2f(dst[t][i]);
      meanacc[t] += wsArr[sc] * (s * (1.f / NN));
    }
    __syncthreads();
    __bf16 (*tmp)[KST] = src; src = dst; dst = tmp;
  }

  // combined = [spectral | ms]
  if (t < DD)            comb[t] = spectral[(size_t)b * DD + t];
  else if (t < HIDN)     comb[t] = meanacc[t - DD];
  __syncthreads();

  // h = relu(combined @ out_w1 + out_b1)
  if (t < HIDN) {
    float acc = ob1[t];
    for (int i = 0; i < HIDN; ++i) acc += comb[i] * ow1[i * HIDN + t];
    hbuf[t] = fmaxf(acc, 0.f);
  }
  __syncthreads();

  // z_e = h @ out_w2 + out_b2
  if (t < NEVK) {
    float acc = ob2[t];
    for (int j = 0; j < HIDN; ++j) acc += hbuf[j] * ow2[j * NEVK + t];
    zebuf[t] = acc;
    out_ze[(size_t)b * NEVK + t] = acc;
  }
  __syncthreads();

  // hyperbolic clip
  float cva = 1.f / (1.f + expf(-curv[0]));      // CURV=1.0 * sigmoid
  if (t == 0) {
    float nrm = 0.f;
    for (int k = 0; k < NEVK; ++k) nrm += zebuf[k] * zebuf[k];
    zebuf[NEVK] = sqrtf(nrm);
  }
  __syncthreads();
  if (t < NEVK) {
    float maxr  = 0.95f * rsqrtf(cva);
    float scale = fminf(1.f, maxr / (zebuf[NEVK] + 1e-12f));
    out_z[(size_t)b * NEVK + t] = zebuf[t] * scale;
  }
  if (b == 0 && t == 0) out_c[0] = cva;
}

// =====================================================================
extern "C" void kernel_launch(void* const* d_in, const int* in_sizes, int n_in,
                              void* d_out, int out_size, void* d_ws, size_t ws_size,
                              hipStream_t stream) {
  (void)n_in; (void)out_size; (void)d_ws; (void)ws_size;
  const float* adj   = (const float*)d_in[0];
  const float* nf    = (const float*)d_in[1];
  const float* sw    = (const float*)d_in[2];
  const float* curv  = (const float*)d_in[3];
  const float* sp_w1 = (const float*)d_in[4];
  const float* sp_b1 = (const float*)d_in[5];
  const float* sp_w2 = (const float*)d_in[6];
  const float* sp_b2 = (const float*)d_in[7];
  const float* ow1   = (const float*)d_in[8];
  const float* ob1   = (const float*)d_in[9];
  const float* ow2   = (const float*)d_in[10];
  const float* ob2   = (const float*)d_in[11];

  const int Bn = in_sizes[0] / (NN * NN);

  float* out  = (float*)d_out;
  float* z    = out;                                       // B*16
  float* ze   = out + (size_t)Bn * NEVK;                   // B*16
  float* spec = out + (size_t)Bn * NEVK * 2;               // B*64
  float* sv   = spec + (size_t)Bn * DD;                    // B*16
  float* cc   = sv + (size_t)Bn * NEVK;                    // 1

  holo_eig_spectral<<<Bn, 256, 0, stream>>>(adj, sp_w1, sp_b1, sp_w2, sp_b2, spec, sv);
  holo_prop_out<<<Bn, 256, 0, stream>>>(adj, nf, sw, curv, ow1, ob1, ow2, ob2,
                                        spec, z, ze, cc);
}
